// CQAttention_66022237274311
// MI455X (gfx1250) — compile-verified, matrix operand order checked by reading, not run on hardware
//
#include <hip/hip_runtime.h>

// ---------------- CQAttention for MI455X (gfx1250, wave32, bf16 WMMA) ----------------
// B=32, Lc=1024, Lq=128, d=256. Memory-bound (~170MB HBM floor ~7us); all intermediates
// (~74MB) are L2-resident (192MB). bf16 WMMA (16x16x32) makes all four GEMMs ~1.7us of
// math vs ~14us for the fp32 16x16x4 path -> bf16 inputs, f32 accumulate, fp32 softmax.
// B-operand panels that are re-read 8-16x per workgroup are staged once into LDS with
// GLOBAL_LOAD_ASYNC_TO_LDS_B128 (ASYNCcnt) and consumed via ds_load_b128.

#define NEG_INF_F (-1e30f)

typedef __attribute__((ext_vector_type(16))) __bf16 v16bf;
typedef __attribute__((ext_vector_type(8)))  float  v8f;

constexpr int B_   = 32;
constexpr int LC   = 1024;
constexpr int LQ   = 128;
constexpr int DIM  = 256;
constexpr int OUTD = 4 * DIM;

struct alignas(32) FragU { unsigned short u[16]; };

__device__ __forceinline__ unsigned short f2bf(float x) {
  union { float f; unsigned int u; } v; v.f = x;
  unsigned int r = v.u + 0x7FFFu + ((v.u >> 16) & 1u);   // round-to-nearest-even
  return (unsigned short)(r >> 16);
}

// Async bulk copy: global -> LDS, 16 bytes per lane per call. No VGPR data; tracked by
// ASYNCcnt. LDS byte address = low 32 bits of the flat shared pointer (LDS aperture).
__device__ __forceinline__ void async_copy_b128(const unsigned short* gsrc,
                                                unsigned short* ldst) {
  unsigned lds = (unsigned)(size_t)ldst;
  asm volatile("global_load_async_to_lds_b128 %0, %1, off"
               :: "v"(lds), "v"(gsrc)
               : "memory");
}
__device__ __forceinline__ void wait_async_all() {
  asm volatile("s_wait_asynccnt 0x0" ::: "memory");
}

// Load one WMMA operand fragment from a row-major [rows x K] bf16 matrix (global or LDS).
// Lane layout per CDNA5 ISA 7.12.2 (16-bit A 16x32): lanes 0-15 row=lane,
// K in {kb..kb+7, kb+16..kb+23}; lanes 16-31 row=lane-16, K in {kb+8.., kb+24..}.
// B operand mirrors this with "row" = output column (i.e. D = X * Y^T).
__device__ __forceinline__ v16bf load_frag(const unsigned short* base,
                                           int row, int ld, int kb, int lane) {
  const int lrow = lane & 15;
  const int hi   = (lane >> 4) & 1;
  const unsigned short* p = base + (size_t)(row + lrow) * ld + kb + hi * 8;
  FragU f;
  *(uint4*)(&f.u[0]) = *(const uint4*)(p);        // K = kb + hi*8 .. +7
  *(uint4*)(&f.u[8]) = *(const uint4*)(p + 16);   // K = kb + 16 + hi*8 .. +7
  return __builtin_bit_cast(v16bf, f);
}

__device__ __forceinline__ v8f wmma_bf16(v16bf a, v16bf b, v8f c) {
  return __builtin_amdgcn_wmma_f32_16x16x32_bf16(false, a, false, b, (short)0, c, false, false);
}

// ---------------- prep: bf16 casts, transposes, row/col linear terms ----------------
__global__ void k_prep_c(const float* __restrict__ C, const float* __restrict__ Wo_w,
                         unsigned short* __restrict__ Ch3, unsigned short* __restrict__ Ct,
                         float* __restrict__ rterm) {
  __shared__ float red[DIM];
  const int idx = blockIdx.x;            // b*LC + c
  const int b = idx / LC, c = idx % LC;
  const int t = threadIdx.x;             // d index
  const float cv = C[(size_t)idx * DIM + t];
  Ch3[(size_t)idx * DIM + t] = f2bf(cv * Wo_w[2 * DIM + t]);     // C * w3 (bf16)
  Ct[((size_t)b * DIM + t) * LC + c] = f2bf(cv);                  // C^T (bf16)
  red[t] = cv * Wo_w[t];                                          // C . w1
  __syncthreads();
  for (int s = DIM / 2; s > 0; s >>= 1) { if (t < s) red[t] += red[t + s]; __syncthreads(); }
  if (t == 0) rterm[idx] = red[0];
}

__global__ void k_prep_q(const float* __restrict__ Q, const float* __restrict__ Wo_w,
                         unsigned short* __restrict__ Qh, unsigned short* __restrict__ Qt,
                         float* __restrict__ qterm) {
  __shared__ float red[DIM];
  const int idx = blockIdx.x;            // b*LQ + q
  const int b = idx / LQ, q = idx % LQ;
  const int t = threadIdx.x;             // d index
  const float qv = Q[(size_t)idx * DIM + t];
  Qh[(size_t)idx * DIM + t] = f2bf(qv);                           // Q (bf16)
  Qt[((size_t)b * DIM + t) * LQ + q] = f2bf(qv);                  // Q^T (bf16)
  red[t] = qv * Wo_w[DIM + t];                                    // Q . w2
  __syncthreads();
  for (int s = DIM / 2; s > 0; s >>= 1) { if (t < s) red[t] += red[t + s]; __syncthreads(); }
  if (t == 0) qterm[idx] = red[0];
}

// ---------------- S = (C*w3) @ Q^T + rterm + qterm + bias  (fp32 logits) ----------------
// The 64KB Q panel is staged once per workgroup into LDS (async), then re-read 8x/wave
// as ds_load_b128 instead of 8x from L2.
__global__ void k_sgemm(const unsigned short* __restrict__ Ch3, const unsigned short* __restrict__ Qh,
                        const float* __restrict__ rterm, const float* __restrict__ qterm,
                        const float* __restrict__ Wo_b, float* __restrict__ S) {
  __shared__ unsigned short sQ[LQ * DIM];            // 64 KB
  const int b = blockIdx.y;
  const int ctile = blockIdx.x * 128;
  const int wave = threadIdx.x >> 5;
  const int lane = threadIdx.x & 31;
  const int rowBase = ctile + wave * 16;              // 8 waves x 16 rows = 128-row tile
  const unsigned short* Xb = Ch3 + (size_t)b * LC * DIM;
  const unsigned short* Yb = Qh  + (size_t)b * LQ * DIM;
  float* Sb = S + (size_t)b * LC * LQ;
  const float bias = Wo_b[0];
  const int lrow = lane & 15, hi = (lane >> 4) & 1;

  for (int i = threadIdx.x * 8; i < LQ * DIM; i += 256 * 8)
    async_copy_b128(Yb + i, &sQ[i]);
  wait_async_all();
  __syncthreads();

  for (int n = 0; n < LQ / 16; ++n) {
    const int colBase = n * 16;
    v8f acc = {};
    for (int kb = 0; kb < DIM; kb += 32) {
      v16bf a = load_frag(Xb, rowBase, DIM, kb, lane);
      v16bf q = load_frag(sQ, colBase, DIM, kb, lane);
      acc = wmma_bf16(a, q, acc);
    }
    for (int i = 0; i < 8; ++i) {
      const int r = rowBase + i + hi * 8;
      const int c = colBase + lrow;
      Sb[(size_t)r * LQ + c] = acc[i] + rterm[b * LC + r] + qterm[b * LQ + c] + bias;
    }
  }
}

// ---------------- softmax over q (rows of S), masked by qmask -> Sq (bf16) ----------------
__global__ void k_softmax_q(const float* __restrict__ S, const int* __restrict__ qmask,
                            unsigned short* __restrict__ SqH) {
  __shared__ float red[LQ];
  const int idx = blockIdx.x;            // b*LC + c
  const int b = idx / LC;
  const int q = threadIdx.x;
  const float m  = (float)qmask[b * LQ + q];
  const float x  = S[(size_t)idx * LQ + q];
  const float xm = x * m + (1.0f - m) * NEG_INF_F;
  red[q] = xm; __syncthreads();
  for (int s = LQ / 2; s > 0; s >>= 1) { if (q < s) red[q] = fmaxf(red[q], red[q + s]); __syncthreads(); }
  const float mx = red[0]; __syncthreads();
  const float e = __expf(xm - mx);
  red[q] = e; __syncthreads();
  for (int s = LQ / 2; s > 0; s >>= 1) { if (q < s) red[q] += red[q + s]; __syncthreads(); }
  SqH[(size_t)idx * LQ + q] = f2bf(e / red[0]);
}

// ------- softmax over c (columns of S), masked by cmask -> Sc^T (bf16, q-major rows) -------
__global__ void k_softmax_c(const float* __restrict__ S, const int* __restrict__ cmask,
                            unsigned short* __restrict__ SctH) {
  __shared__ float red[256];
  const int b = blockIdx.x / LQ;
  const int q = blockIdx.x % LQ;
  const int t = threadIdx.x;
  const float* Sb = S + (size_t)b * LC * LQ;
  float xs[4];
  float mx = NEG_INF_F;
  for (int j = 0; j < 4; ++j) {
    const int c = t + 256 * j;
    const float m = (float)cmask[b * LC + c];
    const float x = Sb[(size_t)c * LQ + q];
    xs[j] = x * m + (1.0f - m) * NEG_INF_F;
    mx = fmaxf(mx, xs[j]);
  }
  red[t] = mx; __syncthreads();
  for (int s = 128; s > 0; s >>= 1) { if (t < s) red[t] = fmaxf(red[t], red[t + s]); __syncthreads(); }
  mx = red[0]; __syncthreads();
  float sum = 0.f, es[4];
  for (int j = 0; j < 4; ++j) { es[j] = __expf(xs[j] - mx); sum += es[j]; }
  red[t] = sum; __syncthreads();
  for (int s = 128; s > 0; s >>= 1) { if (t < s) red[t] += red[t + s]; __syncthreads(); }
  const float inv = 1.0f / red[0];
  unsigned short* dst = SctH + ((size_t)b * LQ + q) * LC;
  for (int j = 0; j < 4; ++j) dst[t + 256 * j] = f2bf(es[j] * inv);
}

// ---------------- T = Sc^T @ C  via  D[q,dd] = Sct * Ct^T ; store T^T (bf16) ----------------
// Streaming K=1024: panels too large to stage; keep global path + global_prefetch_b8.
__global__ void k_tgemm(const unsigned short* __restrict__ SctH, const unsigned short* __restrict__ Ct,
                        unsigned short* __restrict__ TtH) {
  const int b = blockIdx.y;
  const int colBase = blockIdx.x * 16;           // dd sub-tile
  const int wave = threadIdx.x >> 5;
  const int lane = threadIdx.x & 31;
  const int rowBase = wave * 16;                 // 8 waves cover q = 0..127
  const unsigned short* X = SctH + (size_t)b * LQ  * LC;   // q  x c
  const unsigned short* Y = Ct   + (size_t)b * DIM * LC;   // dd x c
  v8f acc = {};
  for (int kb = 0; kb < LC; kb += 32) {
    if (kb + 32 < LC) {  // warm next K-tile (global_prefetch_b8)
      __builtin_prefetch(X + (size_t)(rowBase + (lane & 15)) * LC + kb + 32, 0, 3);
      __builtin_prefetch(Y + (size_t)(colBase + (lane & 15)) * LC + kb + 32, 0, 3);
    }
    v16bf a  = load_frag(X, rowBase, LC, kb, lane);
    v16bf bb = load_frag(Y, colBase, LC, kb, lane);
    acc = wmma_bf16(a, bb, acc);
  }
  const int lrow = lane & 15, hi = (lane >> 4) & 1;
  unsigned short* Tt = TtH + (size_t)b * DIM * LQ;
  for (int i = 0; i < 8; ++i) {
    const int q  = rowBase + i + hi * 8;
    const int dd = colBase + lrow;
    Tt[(size_t)dd * LQ + q] = f2bf(acc[i]);      // transposed store: Tt[dd, q]
  }
}

// ------- A = Sq @ Qt^T ; Bm = Sq @ Tt^T ; out = [C, A, C*A, C*Bm]  (fused epilogue) -------
// Qt and Tt panels (64KB each) staged once per workgroup via async copies (128KB LDS,
// well under the 320KB WGP budget), then re-read 16x/wave as ds_load_b128.
__global__ void k_abgemm(const unsigned short* __restrict__ SqH, const unsigned short* __restrict__ QtH,
                         const unsigned short* __restrict__ TtH, const float* __restrict__ C,
                         float* __restrict__ out) {
  __shared__ unsigned short sQt[DIM * LQ];       // 64 KB
  __shared__ unsigned short sTt[DIM * LQ];       // 64 KB
  const int b = blockIdx.y;
  const int ctile = blockIdx.x * 128;
  const int wave = threadIdx.x >> 5;
  const int lane = threadIdx.x & 31;
  const int rowBase = ctile + wave * 16;
  const unsigned short* X  = SqH + (size_t)b * LC  * LQ;   // c  x q
  const unsigned short* Ya = QtH + (size_t)b * DIM * LQ;   // dd x q
  const unsigned short* Yb = TtH + (size_t)b * DIM * LQ;   // dd x q
  const float* Cb = C + (size_t)b * LC * DIM;
  float* outb = out + (size_t)b * LC * OUTD;
  const int lrow = lane & 15, hi = (lane >> 4) & 1;

  for (int i = threadIdx.x * 8; i < DIM * LQ; i += 256 * 8) {
    async_copy_b128(Ya + i, &sQt[i]);
    async_copy_b128(Yb + i, &sTt[i]);
  }
  wait_async_all();
  __syncthreads();

  for (int n = 0; n < DIM / 16; ++n) {
    const int colBase = n * 16;
    v8f accA = {}, accB = {};
    for (int kb = 0; kb < LQ; kb += 32) {
      v16bf x  = load_frag(X,   rowBase, LQ, kb, lane);    // shared A operand
      v16bf ya = load_frag(sQt, colBase, LQ, kb, lane);
      v16bf yb = load_frag(sTt, colBase, LQ, kb, lane);
      accA = wmma_bf16(x, ya, accA);
      accB = wmma_bf16(x, yb, accB);
    }
    for (int i = 0; i < 8; ++i) {
      const int r  = rowBase + i + hi * 8;
      const int dd = colBase + lrow;
      const float cv = Cb[(size_t)r * DIM + dd];
      float* o = outb + (size_t)r * OUTD;
      o[dd]           = cv;
      o[DIM + dd]     = accA[i];
      o[2 * DIM + dd] = cv * accA[i];
      o[3 * DIM + dd] = cv * accB[i];
    }
  }
}

// ------------------------------------ launcher ------------------------------------
extern "C" void kernel_launch(void* const* d_in, const int* in_sizes, int n_in,
                              void* d_out, int out_size, void* d_ws, size_t ws_size,
                              hipStream_t stream) {
  (void)in_sizes; (void)n_in; (void)out_size; (void)ws_size;
  const float* C     = (const float*)d_in[0];
  const float* Q     = (const float*)d_in[1];
  const int*   cmask = (const int*)d_in[2];
  const int*   qmask = (const int*)d_in[3];
  const float* Wo_w  = (const float*)d_in[4];
  const float* Wo_b  = (const float*)d_in[5];
  float* out = (float*)d_out;

  char* p = (char*)d_ws;
  auto alloc = [&](size_t bytes) { void* r = (void*)p; p += (bytes + 255) & ~(size_t)255; return r; };
  unsigned short* Ch3  = (unsigned short*)alloc((size_t)B_ * LC  * DIM * 2);  // (C*w3) bf16
  unsigned short* Ct   = (unsigned short*)alloc((size_t)B_ * DIM * LC  * 2);  // C^T bf16
  unsigned short* Qh   = (unsigned short*)alloc((size_t)B_ * LQ  * DIM * 2);  // Q bf16
  unsigned short* Qt   = (unsigned short*)alloc((size_t)B_ * DIM * LQ  * 2);  // Q^T bf16
  float*          rterm= (float*)alloc((size_t)B_ * LC * 4);
  float*          qterm= (float*)alloc((size_t)B_ * LQ * 4);
  float*          S    = (float*)alloc((size_t)B_ * LC * LQ * 4);             // logits fp32
  unsigned short* SqH  = (unsigned short*)alloc((size_t)B_ * LC * LQ * 2);    // softmax_q bf16
  unsigned short* SctH = (unsigned short*)alloc((size_t)B_ * LQ * LC * 2);    // softmax_c^T bf16
  unsigned short* TtH  = (unsigned short*)alloc((size_t)B_ * DIM * LQ * 2);   // (Sc^T C)^T bf16

  k_prep_c  <<<dim3(B_ * LC), dim3(DIM), 0, stream>>>(C, Wo_w, Ch3, Ct, rterm);
  k_prep_q  <<<dim3(B_ * LQ), dim3(DIM), 0, stream>>>(Q, Wo_w, Qh, Qt, qterm);
  k_sgemm   <<<dim3(LC / 128, B_), dim3(256), 0, stream>>>(Ch3, Qh, rterm, qterm, Wo_b, S);
  k_softmax_q<<<dim3(B_ * LC), dim3(LQ), 0, stream>>>(S, qmask, SqH);
  k_softmax_c<<<dim3(B_ * LQ), dim3(256), 0, stream>>>(S, cmask, SctH);
  k_tgemm   <<<dim3(DIM / 16, B_), dim3(256), 0, stream>>>(SctH, Ct, TtH);
  k_abgemm  <<<dim3(LC / 128, B_), dim3(256), 0, stream>>>(SqH, Qt, TtH, C, out);
}
